// GHN_53901839565109
// MI455X (gfx1250) — compile-verified, matrix operand order checked by reading, not run on hardware
//
#include <hip/hip_runtime.h>
#include <hip/hip_bf16.h>
#include <math.h>

// ---------------------------------------------------------------------------
// GHN forward for MI455X (gfx1250, wave32, WMMA).
// All dense algebra routed through v_wmma_f32_16x16x32_f16, with LDS-staged
// A fragments and split-K (atomic f32 accumulate) for the long-K encoders.
// ---------------------------------------------------------------------------

typedef __attribute__((ext_vector_type(16))) _Float16 v16h;
typedef __attribute__((ext_vector_type(8)))  float    v8f;

#define HID 64
#define DD  128
#define NN  2048
#define NC  500
#define EE  16384
#define CONV_NUMEL 147456   // 128*128*3*3
#define LIN_NUMEL  16384    // 128*128

// ---------------------------------------------------------------------------
// Generic WMMA GEMM: C[M,N] (+)= A[M,K_chunk] * B[K_chunk,N]
//   accum==0 : C = epilogue(acc + bias)  (scale, optional tanh)
//   accum==1 : C += acc via global_atomic_add_f32 (split-K partial; bias and
//              scale applied later by gemm_epilogue)
// A, B are f32 in memory, converted to f16 in registers (f32 accumulate).
// Block = 128 threads = 4 waves; waves share one M tile, cover 4 N tiles.
// The 16x32 A slab is staged through LDS once per block per K step.
// Requires: N % 64 == 0, K % 32 == 0 (true for every GEMM here).
// ---------------------------------------------------------------------------
__global__ __launch_bounds__(128)
void wmma_gemm_f16(const float* __restrict__ A, int lda,
                   const float* __restrict__ B, int ldb,
                   const float* __restrict__ bias,
                   float* __restrict__ C, int ldc,
                   int M, int N, int K, int kchunk,
                   float scale, int act, int accum)
{
    __shared__ _Float16 As[16 * 32];     // one 16x32 f16 A slab per block

    const int tid  = threadIdx.x;
    const int wave = tid >> 5;
    const int lane = tid & 31;
    const int half = lane >> 4;          // 0: lanes 0-15, 1: lanes 16-31
    const int l16  = lane & 15;

    const int ntN   = N >> 4;
    int tileN       = blockIdx.x * 4 + wave;
    const bool nok  = tileN < ntN;       // store guard (no early exit: barriers)
    tileN           = nok ? tileN : (ntN - 1);
    const int tileM = blockIdx.y;

    const int kbegin = blockIdx.z * kchunk;
    int kend = kbegin + kchunk; if (kend > K) kend = K;

    // cooperative A-slab load coords: 16 rows x 32 k; 8 threads per row, float4 each
    const int arow  = tid >> 3;                    // 0..15
    const int akoff = (tid & 7) * 4;               // 0,4,...,28
    int gRow = tileM * 16 + arow;
    gRow = (gRow < M) ? gRow : (M - 1);            // clamp: edge tiles read valid mem
    const float* __restrict__ Aslab = A + (size_t)gRow * lda;

    const int col = tileN * 16 + l16;
    const float* __restrict__ Bcol = B + col;

    v8f c = {};

    for (int kk = kbegin; kk < kend; kk += 32) {
        // ---- stage A slab: global f32 -> LDS f16 (shared by all 4 waves) ----
        const float4 f4 = *(const float4*)(Aslab + kk + akoff);
        As[arow * 32 + akoff + 0] = (_Float16)f4.x;
        As[arow * 32 + akoff + 1] = (_Float16)f4.y;
        As[arow * 32 + akoff + 2] = (_Float16)f4.z;
        As[arow * 32 + akoff + 3] = (_Float16)f4.w;
        __syncthreads();

        if (kk + 32 < kend) {  // speculative prefetch of next slabs (global_prefetch_b8)
            __builtin_prefetch(Aslab + kk + 32, 0, 0);
            __builtin_prefetch(Bcol + (size_t)(kk + 32) * ldb, 0, 0);
        }

        // ---- A fragment from LDS: ISA 7.12.2 16-bit A 16x32 layout ----
        v16h a;
        #pragma unroll
        for (int v = 0; v < 8; ++v) {
            const int k0 = ((v >= 4) ? 16 : 0) + 2 * (v & 3) + half * 8;
            a[2 * v]     = As[l16 * 32 + k0];
            a[2 * v + 1] = As[l16 * 32 + k0 + 1];
        }

        // ---- B fragment from global: 32x16, K striped like A's K map ----
        v16h b;
        #pragma unroll
        for (int v = 0; v < 8; ++v) {
            const int k0 = ((v >= 4) ? 16 : 0) + 2 * (v & 3) + half * 8;
            b[2 * v]     = (_Float16)Bcol[(size_t)(kk + k0)     * ldb];
            b[2 * v + 1] = (_Float16)Bcol[(size_t)(kk + k0 + 1) * ldb];
        }

        c = __builtin_amdgcn_wmma_f32_16x16x32_f16(
                /*neg_a=*/false, a, /*neg_b=*/false, b,
                /*c_mod=*/(short)0, c, /*reuse_a=*/false, /*reuse_b=*/false);

        __syncthreads();   // protect As before next iteration overwrites it
    }

    // ---- epilogue: C/D layout -> VGPR r holds row (half*8 + r) ----
    if (accum) {
        #pragma unroll
        for (int r = 0; r < 8; ++r) {
            const int row = tileM * 16 + half * 8 + r;
            if (nok && row < M)
                atomicAdd(&C[(size_t)row * ldc + col], c[r]);
        }
    } else {
        const float bv = bias ? bias[col] : 0.0f;
        #pragma unroll
        for (int r = 0; r < 8; ++r) {
            const int row = tileM * 16 + half * 8 + r;
            if (nok && row < M) {
                float t = scale * (c[r] + bv);
                if (act) t = tanhf(t);
                C[(size_t)row * ldc + col] = t;
            }
        }
    }
}

// split-K finishing pass: C = scale*(C + bias[col])  (+ optional tanh)
__global__ void gemm_epilogue(float* __restrict__ C, const float* __restrict__ bias,
                              int total, int N, float scale, int act) {
    int i = blockIdx.x * blockDim.x + threadIdx.x;
    if (i >= total) return;
    float t = scale * (C[i] + bias[i % N]);
    C[i] = act ? tanhf(t) : t;
}

// ---------------------------------------------------------------------------
// Elementwise / scatter-gather helpers
// ---------------------------------------------------------------------------
__global__ void zero_f32(float* __restrict__ p, int n) {
    int i = blockIdx.x * blockDim.x + threadIdx.x;
    if (i < n) p[i] = 0.0f;
}

// feats init: row0 = 1.0 everywhere; rows>=1: cols[64:128) = layer_embed[prim]
__global__ void init_feats(float* __restrict__ h,
                           const int* __restrict__ prim_ids,
                           const float* __restrict__ layer_embed, int n) {
    int idx = blockIdx.x * blockDim.x + threadIdx.x;
    if (idx >= n) return;
    int i = idx >> 7, cc = idx & 127;
    float v;
    if (i == 0)        v = 1.0f;
    else if (cc >= 64) v = layer_embed[prim_ids[i] * HID + (cc - 64)];
    else               v = 0.0f;
    h[idx] = v;
}

// h[idx[r], 0:64) = enc[r, 0:64)
__global__ void scatter_enc(float* __restrict__ h, const int* __restrict__ idx,
                            const float* __restrict__ enc, int n) {
    int t = blockIdx.x * blockDim.x + threadIdx.x;
    if (t >= n) return;
    int r = t >> 6, cc = t & 63;
    h[(size_t)idx[r] * DD + cc] = enc[t];
}

// out[r, :] = h[idx[r], :]   (128 cols)
__global__ void gather_rows(const float* __restrict__ h, const int* __restrict__ idx,
                            float* __restrict__ out, int n) {
    int t = blockIdx.x * blockDim.x + threadIdx.x;
    if (t >= n) return;
    int r = t >> 7, cc = t & 127;
    out[t] = h[(size_t)idx[r] * DD + cc];
}

__global__ void edge_degree(const int* __restrict__ edges,
                            float* __restrict__ din, float* __restrict__ dout, int e) {
    int i = blockIdx.x * blockDim.x + threadIdx.x;
    if (i >= e) return;
    atomicAdd(&din[edges[2 * i + 1]], 1.0f);   // deg_in over dst
    atomicAdd(&dout[edges[2 * i]], 1.0f);      // deg_out over src
}

// aggf[dst] += h[src];  aggb[src] += h[dst]   (per channel)
__global__ void edge_aggregate(const int* __restrict__ edges,
                               const float* __restrict__ h,
                               float* __restrict__ aggf, float* __restrict__ aggb, int n) {
    int t = blockIdx.x * blockDim.x + threadIdx.x;
    if (t >= n) return;
    int e = t >> 7, cc = t & 127;
    int src = edges[2 * e], dst = edges[2 * e + 1];
    atomicAdd(&aggf[(size_t)dst * DD + cc], h[(size_t)src * DD + cc]);
    atomicAdd(&aggb[(size_t)src * DD + cc], h[(size_t)dst * DD + cc]);
}

// m = m1/max(deg_in,1) + m2/max(deg_out,1)
__global__ void combine_msgs(const float* __restrict__ m1, const float* __restrict__ m2,
                             const float* __restrict__ din, const float* __restrict__ dout,
                             float* __restrict__ m, int n) {
    int idx = blockIdx.x * blockDim.x + threadIdx.x;
    if (idx >= n) return;
    int i = idx >> 7;
    m[idx] = m1[idx] / fmaxf(din[i], 1.0f) + m2[idx] / fmaxf(dout[i], 1.0f);
}

// GRU gate fusion: gi,gh are (N, 3*D)
__global__ void gru_update(const float* __restrict__ gi, const float* __restrict__ gh,
                           const float* __restrict__ h, float* __restrict__ hn, int n) {
    int idx = blockIdx.x * blockDim.x + threadIdx.x;
    if (idx >= n) return;
    int i = idx >> 7, cc = idx & 127;
    size_t base = (size_t)i * (3 * DD) + cc;
    float ir = gi[base], iz = gi[base + DD], in_ = gi[base + 2 * DD];
    float hr = gh[base], hz = gh[base + DD], hn_ = gh[base + 2 * DD];
    float r = 1.0f / (1.0f + expf(-(ir + hr)));
    float z = 1.0f / (1.0f + expf(-(iz + hz)));
    float nv = tanhf(in_ + r * hn_);
    hn[idx] = (1.0f - z) * nv + z * h[idx];
}

// ---------------------------------------------------------------------------
// Host-side orchestration
// ---------------------------------------------------------------------------
static inline void launch_gemm(const float* A, int lda, const float* B, int ldb,
                               const float* bias, float* C, int ldc,
                               int M, int N, int K, int splitk,
                               float scale, int act, hipStream_t s) {
    int kchunk = (K + splitk - 1) / splitk;
    kchunk = (kchunk + 31) & ~31;                       // multiple of 32
    const unsigned gz = (unsigned)((K + kchunk - 1) / kchunk);
    dim3 grid((unsigned)(((N >> 4) + 3) / 4), (unsigned)((M + 15) / 16), gz);
    wmma_gemm_f16<<<grid, 128, 0, s>>>(A, lda, B, ldb, bias, C, ldc,
                                       M, N, K, kchunk, scale, act, (int)(gz > 1));
}

static inline unsigned blocks(int n, int t) { return (unsigned)((n + t - 1) / t); }

extern "C" void kernel_launch(void* const* d_in, const int* in_sizes, int n_in,
                              void* d_out, int out_size, void* d_ws, size_t ws_size,
                              hipStream_t stream) {
    (void)in_sizes; (void)n_in; (void)out_size; (void)ws_size;

    // inputs, setup_inputs() dict order (params flattened in insertion order)
    const float* conv_w     = (const float*)d_in[0];   // 500 x 147456
    const float* lin_w      = (const float*)d_in[1];   // 500 x 16384
    const float* bias_w     = (const float*)d_in[2];   // 500 x 128
    const int*   conv_idx   = (const int*)  d_in[3];
    const int*   lin_idx    = (const int*)  d_in[4];
    const int*   bias_idx   = (const int*)  d_in[5];
    const int*   prim_ids   = (const int*)  d_in[6];
    const int*   edges      = (const int*)  d_in[7];   // E x 2
    const float* conv_enc_w = (const float*)d_in[8];   // 147456 x 64
    const float* conv_enc_b = (const float*)d_in[9];
    const float* lin_enc_w  = (const float*)d_in[10];  // 16384 x 64
    const float* lin_enc_b  = (const float*)d_in[11];
    const float* bias_enc_w = (const float*)d_in[12];  // 128 x 64
    const float* bias_enc_b = (const float*)d_in[13];
    const float* layer_emb  = (const float*)d_in[14];  // 16 x 64
    const float* msg_fwd_w  = (const float*)d_in[15];  // 128 x 128
    const float* msg_bwd_w  = (const float*)d_in[16];
    const float* gru_wi     = (const float*)d_in[17];  // 128 x 384
    const float* gru_wh     = (const float*)d_in[18];
    const float* gru_bi     = (const float*)d_in[19];
    const float* gru_bh     = (const float*)d_in[20];
    const float* conv_dec_w = (const float*)d_in[21];  // 128 x 147456
    const float* conv_dec_b = (const float*)d_in[22];
    const float* lin_dec_w  = (const float*)d_in[23];  // 128 x 16384
    const float* lin_dec_b  = (const float*)d_in[24];
    const float* bias_dec_w = (const float*)d_in[25];  // 128 x 128
    const float* bias_dec_b = (const float*)d_in[26];

    // output layout: conv_out | lin_out | bias_out
    float* out_conv = (float*)d_out;
    float* out_lin  = out_conv + (size_t)NC * CONV_NUMEL;
    float* out_bias = out_lin  + (size_t)NC * LIN_NUMEL;

    // workspace layout (floats)
    float* ws    = (float*)d_ws;
    float* enc_c = ws;                    // 500*64  (enc_c, enc_l contiguous)
    float* enc_l = enc_c + NC * HID;
    float* enc_b = enc_l + NC * HID;
    float* h0    = enc_b + NC * HID;      // 2048*128
    float* h1    = h0 + NN * DD;
    float* aggf  = h1 + NN * DD;          // aggf+aggb contiguous (joint zero)
    float* aggb  = aggf + NN * DD;
    float* m1    = aggb + NN * DD;
    float* m2    = m1 + NN * DD;
    float* mm    = m2 + NN * DD;
    float* gi    = mm + NN * DD;          // 2048*384
    float* gh    = gi + NN * 3 * DD;
    float* din   = gh + NN * 3 * DD;      // 2048, din+dout contiguous
    float* dout  = din + NN;
    float* xc    = dout + NN;             // 500*128
    float* xl    = xc + NC * DD;
    float* xb    = xl + NC * DD;

    const int T256 = 256;

    // ---- encoders ----
    // Long-K streams: split-K so thousands of waves stream disjoint conv_w slabs
    // (295 MB @ 23.3 TB/s needs far more concurrency than 32 blocks).
    zero_f32<<<blocks(2 * NC * HID, T256), T256, 0, stream>>>(enc_c, 2 * NC * HID);
    launch_gemm(conv_w, CONV_NUMEL, conv_enc_w, HID, nullptr,
                enc_c, HID, NC, HID, CONV_NUMEL, /*splitk=*/96, 1.0f, 0, stream);
    launch_gemm(lin_w, LIN_NUMEL, lin_enc_w, HID, nullptr,
                enc_l, HID, NC, HID, LIN_NUMEL, /*splitk=*/16, 1.0f, 0, stream);
    gemm_epilogue<<<blocks(NC * HID, T256), T256, 0, stream>>>(enc_c, conv_enc_b,
                                                               NC * HID, HID, 1.0f, 0);
    gemm_epilogue<<<blocks(NC * HID, T256), T256, 0, stream>>>(enc_l, lin_enc_b,
                                                               NC * HID, HID, 1.0f, 0);
    launch_gemm(bias_w, DD, bias_enc_w, HID, bias_enc_b,
                enc_b, HID, NC, HID, DD, 1, 1.0f, 0, stream);

    // ---- node feature assembly ----
    init_feats<<<blocks(NN * DD, T256), T256, 0, stream>>>(h0, prim_ids, layer_emb, NN * DD);
    scatter_enc<<<blocks(NC * HID, T256), T256, 0, stream>>>(h0, conv_idx, enc_c, NC * HID);
    scatter_enc<<<blocks(NC * HID, T256), T256, 0, stream>>>(h0, lin_idx,  enc_l, NC * HID);
    scatter_enc<<<blocks(NC * HID, T256), T256, 0, stream>>>(h0, bias_idx, enc_b, NC * HID);

    // ---- degrees ----
    zero_f32<<<blocks(2 * NN, T256), T256, 0, stream>>>(din, 2 * NN);
    edge_degree<<<blocks(EE, T256), T256, 0, stream>>>(edges, din, dout, EE);

    // ---- T=3 message-passing / GRU steps ----
    float* hc = h0;
    float* hn = h1;
    for (int t = 0; t < 3; ++t) {
        zero_f32<<<blocks(2 * NN * DD, T256), T256, 0, stream>>>(aggf, 2 * NN * DD);
        edge_aggregate<<<blocks(EE * DD, T256), T256, 0, stream>>>(edges, hc, aggf, aggb, EE * DD);
        // segsum commutes with right-multiply: m = (aggf@Wf)/din + (aggb@Wb)/dout
        launch_gemm(aggf, DD, msg_fwd_w, DD, nullptr, m1, DD, NN, DD, DD, 1, 1.0f, 0, stream);
        launch_gemm(aggb, DD, msg_bwd_w, DD, nullptr, m2, DD, NN, DD, DD, 1, 1.0f, 0, stream);
        combine_msgs<<<blocks(NN * DD, T256), T256, 0, stream>>>(m1, m2, din, dout, mm, NN * DD);
        launch_gemm(mm, DD, gru_wi, 3 * DD, gru_bi, gi, 3 * DD, NN, 3 * DD, DD, 1, 1.0f, 0, stream);
        launch_gemm(hc, DD, gru_wh, 3 * DD, gru_bh, gh, 3 * DD, NN, 3 * DD, DD, 1, 1.0f, 0, stream);
        gru_update<<<blocks(NN * DD, T256), T256, 0, stream>>>(gi, gh, hc, hn, NN * DD);
        float* tmp = hc; hc = hn; hn = tmp;
    }

    // ---- decoders ----
    gather_rows<<<blocks(NC * DD, T256), T256, 0, stream>>>(hc, conv_idx, xc, NC * DD);
    gather_rows<<<blocks(NC * DD, T256), T256, 0, stream>>>(hc, lin_idx,  xl, NC * DD);
    gather_rows<<<blocks(NC * DD, T256), T256, 0, stream>>>(hc, bias_idx, xb, NC * DD);

    const float sc_conv = sqrtf(2.0f / 1152.0f);   // 2/(128*3*3)
    const float sc_lin  = sqrtf(2.0f / 128.0f);
    launch_gemm(xc, DD, conv_dec_w, CONV_NUMEL, conv_dec_b,
                out_conv, CONV_NUMEL, NC, CONV_NUMEL, DD, 1, sc_conv, 0, stream);
    launch_gemm(xl, DD, lin_dec_w, LIN_NUMEL, lin_dec_b,
                out_lin, LIN_NUMEL, NC, LIN_NUMEL, DD, 1, sc_lin, 0, stream);
    launch_gemm(xb, DD, bias_dec_w, DD, bias_dec_b,
                out_bias, DD, NC, DD, DD, 1, 0.2f, 1, stream);   // tanh(0.2*(x@W+b))
}
